// GraphTransformer_57732950392871
// MI455X (gfx1250) — compile-verified, matrix operand order checked by reading, not run on hardware
//
#include <hip/hip_runtime.h>
#include <hip/hip_bf16.h>

// ---------------- problem constants (match reference) ----------------
#define NN      4096
#define IN_DIM  256
#define HIDDEN  512
#define CLASSES 256
#define HEADS   4
#define HD      64
#define EDGES   131072
#define ETOT    (EDGES + NN)          // edges + self loops
#define QKVDIM  768                   // 3*IN_DIM

// ---------------- WMMA fragment types ----------------
typedef __attribute__((ext_vector_type(16))) __bf16 v16bf;
typedef __attribute__((ext_vector_type(8)))  float  v8f;
typedef __attribute__((ext_vector_type(4)))  unsigned int u32x4;
typedef __attribute__((ext_vector_type(8)))  int    i32x8;
typedef __attribute__((ext_vector_type(4)))  int    i32x4;

union AFrag { v16bf v; unsigned int d[8]; unsigned short u[16]; };
union CFrag { v8f   v; float f[8]; };

// K-index mapping for 16-bit 16x32 A fragments (ISA 7.12.2):
// lanes 0-15 half, VGPR r in 0..3 -> K = 2r (+8 for lanes 16-31);
// VGPR r in 4..7 -> K = 16 + 2(r-4) (+8 for upper half).
__device__ __forceinline__ int kmap(int r, int half) {
    int base = (r < 4) ? 0 : 16;
    int rr   = (r < 4) ? r : r - 4;
    return base + half * 8 + 2 * rr;
}

__device__ __forceinline__ unsigned short f32_to_bf16(float f) {
    unsigned int u = __float_as_uint(f);
    u += 0x7FFFu + ((u >> 16) & 1u);   // round to nearest even
    return (unsigned short)(u >> 16);
}

__device__ __forceinline__ v8f wmma_bf16(v16bf a, v16bf b, v8f c) {
    return __builtin_amdgcn_wmma_f32_16x16x32_bf16(false, a, false, b, (short)0, c,
                                                   false, false);
}

// ---------------- Tensor Data Mover (TDM) support ----------------
#if __has_builtin(__builtin_amdgcn_tensor_load_to_lds) && \
    __has_builtin(__builtin_amdgcn_s_wait_tensorcnt)
#define HAVE_TDM 1
#else
#define HAVE_TDM 0
#endif

#if HAVE_TDM
// Load a 2D bf16 tile [rows=64][k=32] from gsrc (row stride = K elements)
// into LDS at byte offset lds_off. D# layout per CDNA5 ISA 08_async_tensor 8.3/8.4.
__device__ __forceinline__ void tdm_load_b_tile(const unsigned short* gsrc,
                                                unsigned lds_off, int K, int Ntot) {
    unsigned long long ga = (unsigned long long)gsrc;
    u32x4 g0;
    g0.x = 1u;                                   // count=1 (valid descriptor)
    g0.y = lds_off;                              // lds_addr (bytes)
    g0.z = (unsigned)ga;                         // global_addr[31:0]
    g0.w = (unsigned)((ga >> 32) & 0x1FFFFFFu)   // global_addr[56:32]
         | (2u << 30);                           // type = 2 ("image")
    i32x8 g1;
    g1[0] = (int)(1u << 16);                     // data_size = 1 -> 2 bytes
    g1[1] = (int)(((unsigned)K & 0xFFFFu) << 16);            // tensor_dim0[15:0]
    g1[2] = (int)(((unsigned)K >> 16) |
                  (((unsigned)Ntot & 0xFFFFu) << 16));       // dim0 hi | dim1 lo
    g1[3] = (int)(((unsigned)Ntot >> 16) | (32u << 16));     // dim1 hi | tile_dim0=32
    g1[4] = 64;                                  // tile_dim1=64, tile_dim2=0
    g1[5] = K;                                   // tensor_dim0_stride[31:0]
    g1[6] = 0;                                   // stride0 hi | stride1 lo
    g1[7] = 0;
    i32x4 z4 = (i32x4)0;                         // groups 2/3 unused (2D tensor)
#if defined(__clang_major__) && __clang_major__ >= 23
    i32x8 z8 = (i32x8)0;
    __builtin_amdgcn_tensor_load_to_lds(g0, g1, z4, z4, z8, 0);
#else
    __builtin_amdgcn_tensor_load_to_lds(g0, g1, z4, z4, 0);
#endif
}
#endif

// ---------------- small utility kernels ----------------
__global__ void fill_f32(float* p, float v, int n) {
    int i = blockIdx.x * blockDim.x + threadIdx.x;
    if (i < n) p[i] = v;
}

__global__ void conv_f2b(const float* in, unsigned short* out, int n) {
    int i = blockIdx.x * blockDim.x + threadIdx.x;
    if (i < n) out[i] = f32_to_bf16(in[i]);
}

// in: [K,N] row-major f32  ->  out: [N,K] row-major bf16
__global__ void transpose_f2b(const float* in, unsigned short* out, int K, int N) {
    int i = blockIdx.x * blockDim.x + threadIdx.x;
    if (i >= K * N) return;
    int n = i / K, k = i % K;
    out[(size_t)n * K + k] = f32_to_bf16(in[(size_t)k * N + n]);
}

// build V^T: vT[c][key] = qkv_f[key][512 + c], c in [0,256)
__global__ void build_vT(const float* qkv_f, unsigned short* vT) {
    int i = blockIdx.x * blockDim.x + threadIdx.x;
    if (i >= IN_DIM * NN) return;
    int c = i >> 12, key = i & (NN - 1);
    vT[(size_t)c * NN + key] = f32_to_bf16(qkv_f[(size_t)key * QKVDIM + 512 + c]);
}

// degree accumulation over E edges + N self loops
__global__ void deg_accum(const int* ei, float* deg) {
    int i = blockIdx.x * blockDim.x + threadIdx.x;
    if (i >= ETOT) return;
    int dst = (i < EDGES) ? ei[EDGES + i] : (i - EDGES);
    atomicAdd(&deg[dst], 1.0f);
}

__global__ void deg_to_dinv(float* deg) {
    int i = blockIdx.x * blockDim.x + threadIdx.x;
    if (i >= NN) return;
    float d = deg[i];
    deg[i] = (d > 0.0f) ? rsqrtf(d) : 0.0f;
}

// scatter-add aggregation: out[dst] += h[src] * dinv[src]*dinv[dst], F feats, 4/thread
__global__ void scatter_agg(const int* ei, const float* dinv, const float* h,
                            float* out, int F) {
    long long tid = (long long)blockIdx.x * blockDim.x + threadIdx.x;
    int perEdge = F >> 2;
    long long e = tid / perEdge;
    if (e >= ETOT) return;
    int f4 = (int)(tid % perEdge) * 4;
    int src, dst;
    if (e < EDGES) { src = ei[e]; dst = ei[EDGES + e]; }
    else           { src = (int)(e - EDGES); dst = src; }
    float nrm = dinv[src] * dinv[dst];
    const float4 hv = *(const float4*)(h + (size_t)src * F + f4);
    float* o = out + (size_t)dst * F + f4;
    atomicAdd(o + 0, hv.x * nrm);
    atomicAdd(o + 1, hv.y * nrm);
    atomicAdd(o + 2, hv.z * nrm);
    atomicAdd(o + 3, hv.w * nrm);
}

// epilogue: out = act(in + bias); optional f32 and/or bf16 destinations
__global__ void epilogue(const float* in, const float* bias, float* outf,
                         unsigned short* outb, int rows, int cols, int relu) {
    int i = blockIdx.x * blockDim.x + threadIdx.x;
    if (i >= rows * cols) return;
    float v = in[i];
    if (bias) v += bias[i % cols];
    if (relu) v = fmaxf(v, 0.0f);
    if (outf) outf[i] = v;
    if (outb) outb[i] = f32_to_bf16(v);
}

__global__ void add_relu(const float* a, const float* b, float* out, int n) {
    int i = blockIdx.x * blockDim.x + threadIdx.x;
    if (i < n) out[i] = fmaxf(a[i] + b[i], 0.0f);
}

// ---------------- WMMA GEMM: C[M,N] = A[M,K](bf16) * Bt[N,K](bf16)^T + bias ----------
// Block tile 128(M) x 64(N), 128 threads = 4 waves; wave w owns rows [w*32, w*32+32).
// B tile (64n x 32k) staged in LDS by the Tensor Data Mover (double buffered),
// all 4 waves consume it via ds_load; A fragments software-pipelined from global.
__global__ __launch_bounds__(128) void gemm_bf16(const unsigned short* __restrict__ A,
                                                 const unsigned short* __restrict__ Bt,
                                                 float* __restrict__ C,
                                                 const float* __restrict__ bias,
                                                 int M, int N, int K) {
    const int lane = threadIdx.x & 31;
    const int w    = threadIdx.x >> 5;
    const int half = lane >> 4;
    const int rl   = lane & 15;
    const int mBase = blockIdx.y * 128 + w * 32;
    const int nBase = blockIdx.x * 64;

    CFrag acc[2][4];
#pragma unroll
    for (int ms = 0; ms < 2; ms++)
#pragma unroll
        for (int g = 0; g < 4; g++)
#pragma unroll
            for (int v = 0; v < 8; v++) acc[ms][g].f[v] = 0.0f;

    const unsigned short* arow0 = A + (size_t)(mBase + rl) * K;
    const unsigned short* arow1 = A + (size_t)(mBase + 16 + rl) * K;

#if HAVE_TDM
    __shared__ __align__(16) unsigned short Bl[2][64][32];
    const unsigned ldsBase = (unsigned)(unsigned long long)(size_t)&Bl[0][0][0];

    // prologue: stage first B tile, preload first A fragments
    if (w == 0)
        tdm_load_b_tile(Bt + (size_t)nBase * K, ldsBase, K, N);
    AFrag a0, a1;
#pragma unroll
    for (int r = 0; r < 8; r++) {
        int km = kmap(r, half);
        a0.d[r] = *(const unsigned int*)(arow0 + km);
        a1.d[r] = *(const unsigned int*)(arow1 + km);
    }
    if (w == 0) __builtin_amdgcn_s_wait_tensorcnt((short)0);
    __syncthreads();

    int cur = 0;
    for (int k0 = 0; k0 < K; k0 += 32) {
        const int knext = k0 + 32;
        // issue next TDM stage before compute
        if (knext < K && w == 0)
            tdm_load_b_tile(Bt + (size_t)nBase * K + knext,
                            ldsBase + (unsigned)((cur ^ 1) * 4096), K, N);
        // prefetch next A fragments (wait lands next iteration)
        AFrag a0n, a1n;
        if (knext < K) {
#pragma unroll
            for (int r = 0; r < 8; r++) {
                int km = knext + kmap(r, half);
                a0n.d[r] = *(const unsigned int*)(arow0 + km);
                a1n.d[r] = *(const unsigned int*)(arow1 + km);
            }
        }
        // consume LDS-resident B tile: 8 WMMAs per wave per K-step
#pragma unroll
        for (int g = 0; g < 4; g++) {
            AFrag bf_;
            const unsigned short* bp = &Bl[cur][g * 16 + rl][0];
#pragma unroll
            for (int r = 0; r < 8; r++)
                bf_.d[r] = *(const unsigned int*)(bp + kmap(r, half));
            acc[0][g].v = wmma_bf16(a0.v, bf_.v, acc[0][g].v);
            acc[1][g].v = wmma_bf16(a1.v, bf_.v, acc[1][g].v);
        }
        if (w == 0) __builtin_amdgcn_s_wait_tensorcnt((short)0);
        __syncthreads();
        a0 = a0n; a1 = a1n; cur ^= 1;
    }
#else
    // fallback: direct-global B fragments
    for (int k0 = 0; k0 < K; k0 += 32) {
        AFrag a0, a1;
#pragma unroll
        for (int r = 0; r < 8; r++) {
            int km = k0 + kmap(r, half);
            a0.d[r] = *(const unsigned int*)(arow0 + km);
            a1.d[r] = *(const unsigned int*)(arow1 + km);
        }
#pragma unroll
        for (int g = 0; g < 4; g++) {
            AFrag bf_;
            const unsigned short* brow = Bt + (size_t)(nBase + g * 16 + rl) * K + k0;
#pragma unroll
            for (int r = 0; r < 8; r++)
                bf_.d[r] = *(const unsigned int*)(brow + kmap(r, half));
            acc[0][g].v = wmma_bf16(a0.v, bf_.v, acc[0][g].v);
            acc[1][g].v = wmma_bf16(a1.v, bf_.v, acc[1][g].v);
        }
    }
#endif

#pragma unroll
    for (int ms = 0; ms < 2; ms++)
#pragma unroll
        for (int g = 0; g < 4; g++) {
            int col = nBase + g * 16 + rl;
            float bv = bias ? bias[col] : 0.0f;
#pragma unroll
            for (int v = 0; v < 8; v++) {
                int row = mBase + ms * 16 + half * 8 + v;
                C[(size_t)row * N + col] = acc[ms][g].f[v] + bv;
            }
        }
}

// ---------------- flash attention: one wave per (head, 16-query tile) --------------
// qkv (bf16): [N, 768] = [q | k | v]; vT (bf16): [256, N]; o (f32): [N, 256]
__global__ __launch_bounds__(128) void attn_kernel(const unsigned short* __restrict__ qkv,
                                                   const unsigned short* __restrict__ vT,
                                                   float* __restrict__ o) {
    __shared__ __align__(16) unsigned short Pl[4][16][32];   // per-wave P tile

    const int w    = threadIdx.x >> 5;
    const int lane = threadIdx.x & 31;
    const int wave = blockIdx.x * 4 + w;
    const int head = wave >> 8;        // 256 q-tiles per head
    const int qt   = wave & 255;
    const int half = lane >> 4;
    const int rl   = lane & 15;
    const float scale = 0.125f;        // 1/sqrt(64)

    // resident Q fragments (16 rows x 64 dims = two K-chunks)
    AFrag qa0, qa1;
    {
        const unsigned short* p =
            qkv + (size_t)(qt * 16 + rl) * QKVDIM + head * HD;
#pragma unroll
        for (int r = 0; r < 8; r++) {
            int km = kmap(r, half);
            qa0.d[r] = *(const unsigned int*)(p + km);
            qa1.d[r] = *(const unsigned int*)(p + 32 + km);
        }
    }

    float mrow[8], lrow[8];
    CFrag O[4];
#pragma unroll
    for (int v = 0; v < 8; v++) { mrow[v] = -3.0e38f; lrow[v] = 0.0f; }
#pragma unroll
    for (int g = 0; g < 4; g++)
#pragma unroll
        for (int v = 0; v < 8; v++) O[g].f[v] = 0.0f;

    for (int jb = 0; jb < NN; jb += 32) {
        // ---- scores S(16q x 32k) = Q . K^T ----
        AFrag b00, b01, b10, b11;
        {
            const unsigned short* k0p =
                qkv + (size_t)(jb + rl) * QKVDIM + IN_DIM + head * HD;
            const unsigned short* k1p =
                qkv + (size_t)(jb + 16 + rl) * QKVDIM + IN_DIM + head * HD;
#pragma unroll
            for (int r = 0; r < 8; r++) {
                int km = kmap(r, half);
                b00.d[r] = *(const unsigned int*)(k0p + km);
                b01.d[r] = *(const unsigned int*)(k0p + 32 + km);
                b10.d[r] = *(const unsigned int*)(k1p + km);
                b11.d[r] = *(const unsigned int*)(k1p + 32 + km);
            }
        }
        CFrag S0, S1;
#pragma unroll
        for (int v = 0; v < 8; v++) { S0.f[v] = 0.0f; S1.f[v] = 0.0f; }
        S0.v = wmma_bf16(qa0.v, b00.v, S0.v);
        S0.v = wmma_bf16(qa1.v, b01.v, S0.v);
        S1.v = wmma_bf16(qa0.v, b10.v, S1.v);
        S1.v = wmma_bf16(qa1.v, b11.v, S1.v);

        // ---- online softmax; rows of C layout live in 16-lane halves ----
#pragma unroll
        for (int v = 0; v < 8; v++) {
            float s0 = S0.f[v] * scale;
            float s1 = S1.f[v] * scale;
            float rm = fmaxf(s0, s1);
#pragma unroll
            for (int m = 8; m >= 1; m >>= 1) rm = fmaxf(rm, __shfl_xor(rm, m, 32));
            float mnew = fmaxf(mrow[v], rm);
            float al   = __expf(mrow[v] - mnew);
            float p0   = __expf(s0 - mnew);
            float p1   = __expf(s1 - mnew);
            float rs   = p0 + p1;
#pragma unroll
            for (int m = 8; m >= 1; m >>= 1) rs += __shfl_xor(rs, m, 32);
            lrow[v] = lrow[v] * al + rs;
            mrow[v] = mnew;
            int row = half * 8 + v;
            Pl[w][row][rl]      = f32_to_bf16(p0);
            Pl[w][row][16 + rl] = f32_to_bf16(p1);
#pragma unroll
            for (int g = 0; g < 4; g++) O[g].f[v] *= al;
        }
        __builtin_amdgcn_wave_barrier();   // LDS re-layout within the wave

        // ---- P as A fragment (C layout -> A layout via LDS) ----
        AFrag pf;
#pragma unroll
        for (int r = 0; r < 8; r++)
            pf.d[r] = *(const unsigned int*)&Pl[w][rl][kmap(r, half)];
        __builtin_amdgcn_wave_barrier();

        // ---- O += P(16x32) . V(32x64), 4 output column groups ----
#pragma unroll
        for (int g = 0; g < 4; g++) {
            AFrag vb;
            const unsigned short* vp =
                vT + (size_t)(head * HD + g * 16 + rl) * NN + jb;
#pragma unroll
            for (int r = 0; r < 8; r++)
                vb.d[r] = *(const unsigned int*)(vp + kmap(r, half));
            O[g].v = wmma_bf16(pf.v, vb.v, O[g].v);
        }
    }

    // ---- normalize and store ----
#pragma unroll
    for (int v = 0; v < 8; v++) {
        float inv = 1.0f / lrow[v];
        int row = qt * 16 + half * 8 + v;
#pragma unroll
        for (int g = 0; g < 4; g++)
            o[(size_t)row * IN_DIM + head * HD + g * 16 + rl] = O[g].f[v] * inv;
    }
}

// ---------------- host orchestration ----------------
static inline int ceil_div(long long a, int b) { return (int)((a + b - 1) / b); }

extern "C" void kernel_launch(void* const* d_in, const int* in_sizes, int n_in,
                              void* d_out, int out_size, void* d_ws, size_t ws_size,
                              hipStream_t stream) {
    (void)in_sizes; (void)n_in; (void)out_size; (void)ws_size;

    const float* x          = (const float*)d_in[0];
    const int*   ei         = (const int*)d_in[1];
    const float* gcn1_w     = (const float*)d_in[2];
    const float* gcn1_b     = (const float*)d_in[3];
    const float* gcn2_w     = (const float*)d_in[4];
    const float* gcn2_b     = (const float*)d_in[5];
    const float* lin_w      = (const float*)d_in[6];
    const float* lin_b      = (const float*)d_in[7];
    const float* in_proj_w  = (const float*)d_in[8];
    const float* in_proj_b  = (const float*)d_in[9];
    const float* out_proj_w = (const float*)d_in[10];
    const float* out_proj_b = (const float*)d_in[11];
    const float* proj_w     = (const float*)d_in[12];
    const float* proj_b     = (const float*)d_in[13];
    float* out = (float*)d_out;

    char* ws = (char*)d_ws;
    size_t off = 0;
    auto alloc = [&](size_t bytes) -> char* {
        char* p = ws + off;
        off = (off + bytes + 255) & ~(size_t)255;
        return p;
    };
    unsigned short* xb     = (unsigned short*)alloc((size_t)NN * IN_DIM * 2);
    unsigned short* w1t    = (unsigned short*)alloc((size_t)HIDDEN * IN_DIM * 2);
    unsigned short* w2t    = (unsigned short*)alloc((size_t)HIDDEN * HIDDEN * 2);
    unsigned short* wlint  = (unsigned short*)alloc((size_t)CLASSES * HIDDEN * 2);
    unsigned short* winpb  = (unsigned short*)alloc((size_t)QKVDIM * IN_DIM * 2);
    unsigned short* woutpb = (unsigned short*)alloc((size_t)IN_DIM * IN_DIM * 2);
    unsigned short* wprojt = (unsigned short*)alloc((size_t)CLASSES * IN_DIM * 2);
    float*          deg    = (float*)alloc((size_t)NN * 4);
    float*          tbuf   = (float*)alloc((size_t)NN * HIDDEN * 4);
    float*          agg    = (float*)alloc((size_t)NN * HIDDEN * 4);
    unsigned short* hb     = (unsigned short*)alloc((size_t)NN * HIDDEN * 2);
    float*          qkv_f  = (float*)alloc((size_t)NN * QKVDIM * 4);
    unsigned short* qkv_b  = (unsigned short*)alloc((size_t)NN * QKVDIM * 2);
    unsigned short* vT     = (unsigned short*)alloc((size_t)IN_DIM * NN * 2);
    float*          o_f    = (float*)alloc((size_t)NN * IN_DIM * 4);
    unsigned short* o_b    = (unsigned short*)alloc((size_t)NN * IN_DIM * 2);
    float*          tf_f   = (float*)alloc((size_t)NN * IN_DIM * 4);
    unsigned short* tf_b   = (unsigned short*)alloc((size_t)NN * IN_DIM * 2);
    float*          proj_f = (float*)alloc((size_t)NN * CLASSES * 4);

    const int T = 256;

    // ---- precision conversion / weight packing ----
    conv_f2b<<<ceil_div((long long)NN * IN_DIM, T), T, 0, stream>>>(x, xb, NN * IN_DIM);
    transpose_f2b<<<ceil_div((long long)IN_DIM * HIDDEN, T), T, 0, stream>>>(gcn1_w, w1t, IN_DIM, HIDDEN);
    transpose_f2b<<<ceil_div((long long)HIDDEN * HIDDEN, T), T, 0, stream>>>(gcn2_w, w2t, HIDDEN, HIDDEN);
    transpose_f2b<<<ceil_div((long long)HIDDEN * CLASSES, T), T, 0, stream>>>(lin_w, wlint, HIDDEN, CLASSES);
    conv_f2b<<<ceil_div((long long)QKVDIM * IN_DIM, T), T, 0, stream>>>(in_proj_w, winpb, QKVDIM * IN_DIM);
    conv_f2b<<<ceil_div((long long)IN_DIM * IN_DIM, T), T, 0, stream>>>(out_proj_w, woutpb, IN_DIM * IN_DIM);
    transpose_f2b<<<ceil_div((long long)IN_DIM * CLASSES, T), T, 0, stream>>>(proj_w, wprojt, IN_DIM, CLASSES);

    // ---- GCN degree normalization ----
    fill_f32<<<ceil_div(NN, T), T, 0, stream>>>(deg, 0.0f, NN);
    deg_accum<<<ceil_div(ETOT, T), T, 0, stream>>>(ei, deg);
    deg_to_dinv<<<ceil_div(NN, T), T, 0, stream>>>(deg);

    // ---- GCN layer 1 ----
    gemm_bf16<<<dim3(HIDDEN / 64, NN / 128), 128, 0, stream>>>(xb, w1t, tbuf, nullptr,
                                                               NN, HIDDEN, IN_DIM);
    fill_f32<<<ceil_div((long long)NN * HIDDEN, T), T, 0, stream>>>(agg, 0.0f, NN * HIDDEN);
    scatter_agg<<<ceil_div((long long)ETOT * (HIDDEN / 4), T), T, 0, stream>>>(ei, deg, tbuf, agg, HIDDEN);
    epilogue<<<ceil_div((long long)NN * HIDDEN, T), T, 0, stream>>>(agg, gcn1_b, nullptr, hb,
                                                                    NN, HIDDEN, 1);
    // ---- GCN layer 2 ----
    gemm_bf16<<<dim3(HIDDEN / 64, NN / 128), 128, 0, stream>>>(hb, w2t, tbuf, nullptr,
                                                               NN, HIDDEN, HIDDEN);
    fill_f32<<<ceil_div((long long)NN * HIDDEN, T), T, 0, stream>>>(agg, 0.0f, NN * HIDDEN);
    scatter_agg<<<ceil_div((long long)ETOT * (HIDDEN / 4), T), T, 0, stream>>>(ei, deg, tbuf, agg, HIDDEN);
    epilogue<<<ceil_div((long long)NN * HIDDEN, T), T, 0, stream>>>(agg, gcn2_b, nullptr, hb,
                                                                    NN, HIDDEN, 1);
    // ---- x_gnn = h @ lin_w + lin_b  -> d_out ----
    gemm_bf16<<<dim3(CLASSES / 64, NN / 128), 128, 0, stream>>>(hb, wlint, out, lin_b,
                                                                NN, CLASSES, HIDDEN);

    // ---- MHA: qkv projection ----
    gemm_bf16<<<dim3(QKVDIM / 64, NN / 128), 128, 0, stream>>>(xb, winpb, qkv_f, in_proj_b,
                                                               NN, QKVDIM, IN_DIM);
    conv_f2b<<<ceil_div((long long)NN * QKVDIM, T), T, 0, stream>>>(qkv_f, qkv_b, NN * QKVDIM);
    build_vT<<<ceil_div((long long)IN_DIM * NN, T), T, 0, stream>>>(qkv_f, vT);

    // ---- flash attention: 4 heads x 256 q-tiles = 1024 waves ----
    attn_kernel<<<HEADS * (NN / 16) / 4, 128, 0, stream>>>(qkv_b, vT, o_f);

    // ---- out projection + final projection ----
    conv_f2b<<<ceil_div((long long)NN * IN_DIM, T), T, 0, stream>>>(o_f, o_b, NN * IN_DIM);
    gemm_bf16<<<dim3(IN_DIM / 64, NN / 128), 128, 0, stream>>>(o_b, woutpb, tf_f, out_proj_b,
                                                               NN, IN_DIM, IN_DIM);
    conv_f2b<<<ceil_div((long long)NN * IN_DIM, T), T, 0, stream>>>(tf_f, tf_b, NN * IN_DIM);
    gemm_bf16<<<dim3(CLASSES / 64, NN / 128), 128, 0, stream>>>(tf_b, wprojt, proj_f, proj_b,
                                                                NN, CLASSES, IN_DIM);

    // ---- out = relu(x_gnn + x_proj) ----
    add_relu<<<ceil_div((long long)NN * CLASSES, T), T, 0, stream>>>(out, proj_f, out,
                                                                     NN * CLASSES);
}
